// GatedDeltaNet_12781822672944
// MI455X (gfx1250) — matrix operand order repaired and checked by reading, not compiled, not run on hardware
//
#include <hip/hip_runtime.h>
#include <hip/hip_bf16.h>

// ---------------------------------------------------------------------------
// Gated DeltaNet forward for MI455X (gfx1250, wave32, WMMA, async-to-LDS)
// B=2 T=1024 D=1536 H=12 Dk=64 Dv=128 K=4  TQK=768 TV=1536
// ---------------------------------------------------------------------------

typedef __bf16 bf16_t;
typedef __attribute__((ext_vector_type(16))) __bf16 v16bf;
typedef __attribute__((ext_vector_type(8)))  float  v8f;
typedef int v4i_gcc __attribute__((vector_size(16)));   // matches builtin's V4i
typedef __attribute__((address_space(1))) v4i_gcc* gas_b128;
typedef __attribute__((address_space(3))) v4i_gcc* las_b128;

#define Bq   2
#define Tq   1024
#define Dq   1536
#define Hq   12
#define DKq  64
#define DVq  128
#define TQK  768
#define TVq  1536
#define NCAT 4608   // 768 + 768 + 1536 + 1536 (q,k,v,g)
#define BT   2048   // B*T

#if __has_builtin(__builtin_amdgcn_global_load_async_to_lds_b128)
#define HAVE_ASYNC_LDS 1
#else
#define HAVE_ASYNC_LDS 0
#endif

__device__ __forceinline__ void wait_async0() {
#if __has_builtin(__builtin_amdgcn_s_wait_asynccnt)
    __builtin_amdgcn_s_wait_asynccnt(0);
#else
    asm volatile("s_wait_asynccnt 0" ::: "memory");
#endif
}

__device__ __forceinline__ float sigmoidf_(float x) { return 1.0f / (1.0f + expf(-x)); }

__device__ __forceinline__ unsigned short f32_to_bf16_rne(float f) {
    union { float f; unsigned int u; } c; c.f = f;
    unsigned int u = c.u;
    u += 0x7fffu + ((u >> 16) & 1u);
    return (unsigned short)(u >> 16);
}

// ---------------------------------------------------------------------------
// f32 -> bf16 convert (RNE)
// ---------------------------------------------------------------------------
__global__ void cvt_f32_bf16(const float* __restrict__ src, bf16_t* __restrict__ dst, int n) {
    int i = blockIdx.x * blockDim.x + threadIdx.x;
    if (i < n) {
        union { unsigned short s; bf16_t b; } cv;
        cv.s = f32_to_bf16_rne(src[i]);
        dst[i] = cv.b;
    }
}

// ---------------------------------------------------------------------------
// GEMM: C[M,N] = A[M,K] * B[N,K]^T   (bf16 in, f32 out) via v_wmma_f32_16x16x32_bf16
// Block tile 128(M) x 128(N), BK=32. 256 threads = 8 waves (4 M-waves x 2 N-waves).
// Each wave computes 32x64 = 8 WMMA tiles (2 A frags, 4 B frags reused 2x).
// Async double-buffered LDS staging (GLOBAL_LOAD_ASYNC_TO_LDS_B128, ASYNCcnt),
// one barrier per K-step.
// ---------------------------------------------------------------------------
#define GBM 128
#define GBN 128
#define GBK 32
#define LDAP 40   // padded LDS row stride (bf16 elems), 80B, keeps 16B alignment
#define LDBP 40

__launch_bounds__(256)
__global__ void gemm_bf16_wmma(const bf16_t* __restrict__ A,
                               const bf16_t* __restrict__ B,
                               float* __restrict__ C,
                               int M, int N, int K) {
    __shared__ __align__(16) bf16_t lA[2][GBM * LDAP];
    __shared__ __align__(16) bf16_t lB[2][GBN * LDBP];

    const int tid  = threadIdx.x;
    const int lane = tid & 31;
    const int wave = tid >> 5;     // 0..7
    const int wm   = wave & 3;     // 0..3  (M direction, 32 rows each)
    const int wn   = wave >> 2;    // 0..1  (N direction, 64 cols each)
    const int m0   = blockIdx.y * GBM;
    const int n0   = blockIdx.x * GBN;

    // staging: 128 rows x 32 bf16 (64B/row) per tile; 2 threads/row x 32B (2x b128)
    const int sr = tid >> 1;            // 0..127
    const int sc = (tid & 1) * 16;      // bf16 offset 0 / 16
    const bf16_t* gA = A + (size_t)(m0 + sr) * K + sc;
    const bf16_t* gB = B + (size_t)(n0 + sr) * K + sc;

    v8f acc[2][4] = {};

#if HAVE_ASYNC_LDS
    auto stage_async = [&](int buf, int kk) {
        __builtin_amdgcn_global_load_async_to_lds_b128(
            (gas_b128)(gA + kk),
            (las_b128)(&lA[buf][sr * LDAP + sc]), 0, 0);
        __builtin_amdgcn_global_load_async_to_lds_b128(
            (gas_b128)(gA + kk + 8),
            (las_b128)(&lA[buf][sr * LDAP + sc + 8]), 0, 0);
        __builtin_amdgcn_global_load_async_to_lds_b128(
            (gas_b128)(gB + kk),
            (las_b128)(&lB[buf][sr * LDBP + sc]), 0, 0);
        __builtin_amdgcn_global_load_async_to_lds_b128(
            (gas_b128)(gB + kk + 8),
            (las_b128)(&lB[buf][sr * LDBP + sc + 8]), 0, 0);
    };
#endif

    auto compute = [&](int buf) {
        union FragA { uint4 u[2]; v16bf v; } af[2];
        #pragma unroll
        for (int sm = 0; sm < 2; ++sm) {
            // A frag (16x32): lane<16 -> row, K {0..7,16..23}; lane>=16 -> K {8..15,24..31}
            int r  = wm * 32 + sm * 16 + (lane & 15);
            int kh = (lane >> 4) * 8;
            af[sm].u[0] = *(const uint4*)(&lA[buf][r * LDAP + kh]);
            af[sm].u[1] = *(const uint4*)(&lA[buf][r * LDAP + 16 + kh]);
        }
        #pragma unroll
        for (int s = 0; s < 4; ++s) {
            // B frag (32x16): lane<16 -> col n, K {0..7,16..23}; lane>=16 -> K {8..15,24..31}
            union FragB { uint4 u[2]; v16bf v; } bfr;
            int r  = wn * 64 + s * 16 + (lane & 15);
            int ko = (lane >> 4) * 8;
            bfr.u[0] = *(const uint4*)(&lB[buf][r * LDBP + ko]);
            bfr.u[1] = *(const uint4*)(&lB[buf][r * LDBP + ko + 16]);
            acc[0][s] = __builtin_amdgcn_wmma_f32_16x16x32_bf16(
                false, af[0].v, false, bfr.v, (short)0, acc[0][s], false, false);
            acc[1][s] = __builtin_amdgcn_wmma_f32_16x16x32_bf16(
                false, af[1].v, false, bfr.v, (short)0, acc[1][s], false, false);
        }
    };

#if HAVE_ASYNC_LDS
    stage_async(0, 0);
    const int nk = K / GBK;
    for (int i = 0; i < nk; ++i) {
        wait_async0();        // this wave's async writes into buf (i&1) have landed
        __syncthreads();      // all waves' writes visible; all waves done reading buf ((i+1)&1)
        if (i + 1 < nk) stage_async((i + 1) & 1, (i + 1) * GBK);
        compute(i & 1);
    }
#else
    for (int kk = 0; kk < K; kk += GBK) {
        __syncthreads();
        uint4 a0 = *(const uint4*)(gA + kk);
        uint4 a1 = *(const uint4*)(gA + kk + 8);
        uint4 b0 = *(const uint4*)(gB + kk);
        uint4 b1 = *(const uint4*)(gB + kk + 8);
        *(uint4*)(&lA[0][sr * LDAP + sc])     = a0;
        *(uint4*)(&lA[0][sr * LDAP + sc + 8]) = a1;
        *(uint4*)(&lB[0][sr * LDBP + sc])     = b0;
        *(uint4*)(&lB[0][sr * LDBP + sc + 8]) = b1;
        __syncthreads();
        compute(0);
    }
#endif

    // store: VGPR r of C/D holds M = r (lanes 0-15) and M = 8+r (lanes 16-31); N = lane&15
    {
        int col   = lane & 15;
        int rbase = (lane >> 4) * 8;
        #pragma unroll
        for (int sm = 0; sm < 2; ++sm) {
            #pragma unroll
            for (int s = 0; s < 4; ++s) {
                int n = n0 + wn * 64 + s * 16 + col;
                #pragma unroll
                for (int r = 0; r < 8; ++r) {
                    C[(size_t)(m0 + wm * 32 + sm * 16 + rbase + r) * N + n] = acc[sm][s][r];
                }
            }
        }
    }
}

// ---------------------------------------------------------------------------
// alpha/beta gates: one wave per (bt, h)
// alpha = exp(-exp(A_log[h]) * softplus(x.w_a + dt_bias[h]));  beta = 2*sigmoid(x.w_b)
// ---------------------------------------------------------------------------
__launch_bounds__(256)
__global__ void gates_kernel(const float* __restrict__ x,
                             const float* __restrict__ wa,
                             const float* __restrict__ wb,
                             const float* __restrict__ A_log,
                             const float* __restrict__ dt_bias,
                             float* __restrict__ alpha,
                             float* __restrict__ beta) {
    int gw   = blockIdx.x * 8 + (threadIdx.x >> 5);  // global wave id, 0..24575
    int lane = threadIdx.x & 31;
    int bt = gw / Hq;
    int h  = gw % Hq;
    const float* xr  = x  + (size_t)bt * Dq;
    const float* war = wa + (size_t)h * Dq;
    const float* wbr = wb + (size_t)h * Dq;
    float da = 0.f, db = 0.f;
    for (int i = lane; i < Dq; i += 32) {
        float xv = xr[i];
        da += xv * war[i];
        db += xv * wbr[i];
    }
    #pragma unroll
    for (int off = 16; off > 0; off >>= 1) {
        da += __shfl_xor(da, off, 32);
        db += __shfl_xor(db, off, 32);
    }
    if (lane == 0) {
        float z  = da + dt_bias[h];
        float sp = (z > 20.f) ? z : log1pf(expf(z));
        alpha[(size_t)bt * Hq + h] = expf(-expf(A_log[h]) * sp);
        beta [(size_t)bt * Hq + h] = 2.f * sigmoidf_(db);
    }
}

// ---------------------------------------------------------------------------
// q/k causal depthwise conv (K=4) + SiLU + per-head L2 norm. One block per (b,t),
// 768 threads (one per channel). Y layout: (BT, 4608), q cols [0,768), k cols [768,1536)
// ---------------------------------------------------------------------------
__launch_bounds__(768)
__global__ void qk_conv_norm(const float* __restrict__ Y,
                             const float* __restrict__ qw,   // (768,1,4) flat
                             const float* __restrict__ kw,
                             float* __restrict__ qn,          // (BT,768)
                             float* __restrict__ kn) {
    __shared__ float wsum_q[24], wsum_k[24];
    int bt = blockIdx.x;
    int t  = bt & (Tq - 1);
    int c  = threadIdx.x;
    float aq = 0.f, ak = 0.f;
    #pragma unroll
    for (int j = 0; j < 4; ++j) {
        int tt = t - 3 + j;
        if (tt >= 0) {
            const float* row = Y + (size_t)(bt - 3 + j) * NCAT;
            aq += qw[c * 4 + j] * row[c];
            ak += kw[c * 4 + j] * row[768 + c];
        }
    }
    aq *= sigmoidf_(aq);   // SiLU
    ak *= sigmoidf_(ak);
    // per-head (64-ch) sum of squares: intra-wave shuffle, then combine 2 waves
    float pq = aq * aq, pk = ak * ak;
    #pragma unroll
    for (int off = 16; off > 0; off >>= 1) {
        pq += __shfl_xor(pq, off, 32);
        pk += __shfl_xor(pk, off, 32);
    }
    int w = c >> 5;
    if ((c & 31) == 0) { wsum_q[w] = pq; wsum_k[w] = pk; }
    __syncthreads();
    int g = (c >> 6) << 1;                 // first wave of this 64-ch group
    float nq = sqrtf(wsum_q[g] + wsum_q[g + 1]);
    float nk = sqrtf(wsum_k[g] + wsum_k[g + 1]);
    qn[(size_t)bt * TQK + c] = aq / fmaxf(nq, 1e-6f);
    kn[(size_t)bt * TQK + c] = ak / fmaxf(nk, 1e-6f);
}

// ---------------------------------------------------------------------------
// v causal depthwise conv + SiLU. One block per (b,t). v cols [1536,3072) of Y.
// ---------------------------------------------------------------------------
__launch_bounds__(512)
__global__ void v_conv(const float* __restrict__ Y,
                       const float* __restrict__ vw,
                       float* __restrict__ vc) {        // (BT,1536)
    int bt = blockIdx.x;
    int t  = bt & (Tq - 1);
    for (int c = threadIdx.x; c < TVq; c += blockDim.x) {
        float a = 0.f;
        #pragma unroll
        for (int j = 0; j < 4; ++j) {
            int tt = t - 3 + j;
            if (tt >= 0) a += vw[c * 4 + j] * Y[(size_t)(bt - 3 + j) * NCAT + 1536 + c];
        }
        a *= sigmoidf_(a);
        vc[(size_t)bt * TVq + c] = a;
    }
}

// ---------------------------------------------------------------------------
// Sequential delta-rule scan. One block per (b,h) = 24 blocks, 128 lanes.
// Lane owns column v of state S (64 f32 in VGPRs).
// S[d,v] = alpha*S[d,v] + k_d*(v_v - alpha*beta*(k^T S)_v);  o_v = sum_d q_d S[d,v]
// Latency-bound (24 blocks): prefetch t+4 operands (global_prefetch_b8).
// ---------------------------------------------------------------------------
__launch_bounds__(128)
__global__ void gdn_scan_kernel(const float* __restrict__ qn,
                                const float* __restrict__ kn,
                                const float* __restrict__ vc,
                                const float* __restrict__ alpha,
                                const float* __restrict__ beta,
                                float* __restrict__ o) {   // (BT,1536)
    int bh = blockIdx.x;
    int b  = bh / Hq;
    int h  = bh % Hq;
    int vcol = threadIdx.x;
    __shared__ float s_k[DKq], s_q[DKq], s_ab[2];
    float S[DKq];
    #pragma unroll
    for (int d = 0; d < DKq; ++d) S[d] = 0.f;

    for (int t = 0; t < Tq; ++t) {
        size_t bt = (size_t)b * Tq + t;
        if (t + 4 < Tq) {
            size_t btp = bt + 4;
            __builtin_prefetch(&vc[btp * TVq + h * DVq + vcol], 0, 1);
            if (vcol < DKq) {
                __builtin_prefetch(&kn[btp * TQK + h * DKq + vcol], 0, 1);
                __builtin_prefetch(&qn[btp * TQK + h * DKq + vcol], 0, 1);
            }
        }
        __syncthreads();
        if (vcol < DKq) {
            s_k[vcol] = kn[bt * TQK + h * DKq + vcol];
            s_q[vcol] = qn[bt * TQK + h * DKq + vcol];
        }
        if (vcol == 0) {
            s_ab[0] = alpha[bt * Hq + h];
            s_ab[1] = beta [bt * Hq + h];
        }
        __syncthreads();
        float vv = vc[bt * TVq + h * DVq + vcol];
        float al = s_ab[0];
        float be = s_ab[1];
        float kS = 0.f;
        #pragma unroll
        for (int d = 0; d < DKq; ++d) kS += s_k[d] * S[d];
        float c1 = vv - al * be * kS;
        float ot = 0.f;
        #pragma unroll
        for (int d = 0; d < DKq; ++d) {
            float sd = al * S[d] + s_k[d] * c1;
            S[d] = sd;
            ot += s_q[d] * sd;
        }
        o[bt * TVq + h * DVq + vcol] = ot;
    }
}

// ---------------------------------------------------------------------------
// RMS norm (per head, Dv=128) + SiLU(g) gate + pack to bf16. One wave per (bt,h).
// g cols [3072,4608) of Y.
// ---------------------------------------------------------------------------
__launch_bounds__(256)
__global__ void rms_gate_kernel(const float* __restrict__ o,
                                const float* __restrict__ Y,
                                bf16_t* __restrict__ og) {   // (BT,1536) bf16
    int gw   = blockIdx.x * 8 + (threadIdx.x >> 5);
    int lane = threadIdx.x & 31;
    int bt = gw / Hq;
    int h  = gw % Hq;
    const float* orow = o + (size_t)bt * TVq + h * DVq;
    const float* grow = Y + (size_t)bt * NCAT + 3072 + h * DVq;
    float vals[4];
    float ss = 0.f;
    #pragma unroll
    for (int i = 0; i < 4; ++i) {
        float v = orow[lane + 32 * i];
        vals[i] = v;
        ss += v * v;
    }
    #pragma unroll
    for (int off = 16; off > 0; off >>= 1) ss += __shfl_xor(ss, off, 32);
    float r = rsqrtf(ss * (1.0f / DVq) + 1e-6f);
    #pragma unroll
    for (int i = 0; i < 4; ++i) {
        float g   = grow[lane + 32 * i];
        float out = vals[i] * r * (g * sigmoidf_(g));
        union { unsigned short s; bf16_t b; } cv;
        cv.s = f32_to_bf16_rne(out);
        og[(size_t)bt * TVq + h * DVq + lane + 32 * i] = cv.b;
    }
}

// ---------------------------------------------------------------------------
// Host-side orchestration
// ---------------------------------------------------------------------------
extern "C" void kernel_launch(void* const* d_in, const int* in_sizes, int n_in,
                              void* d_out, int out_size, void* d_ws, size_t ws_size,
                              hipStream_t stream) {
    const float* x      = (const float*)d_in[0];
    const float* w_q    = (const float*)d_in[1];
    const float* w_k    = (const float*)d_in[2];
    const float* w_v    = (const float*)d_in[3];
    const float* w_a    = (const float*)d_in[4];
    const float* w_b    = (const float*)d_in[5];
    const float* w_g    = (const float*)d_in[6];
    const float* w_out  = (const float*)d_in[7];
    const float* A_log  = (const float*)d_in[8];
    const float* dt_b   = (const float*)d_in[9];
    const float* q_cw   = (const float*)d_in[10];
    const float* k_cw   = (const float*)d_in[11];
    const float* v_cw   = (const float*)d_in[12];
    float* out = (float*)d_out;

    char* ws = (char*)d_ws;
    // workspace layout (bytes)
    bf16_t* xb    = (bf16_t*)(ws + 0);                    //  6,291,456
    bf16_t* Wcat  = (bf16_t*)(ws + 6291456);              // 14,155,776
    float*  Y     = (float*) (ws + 20447232);             // 37,748,736
    float*  qn    = (float*) (ws + 58195968);             //  6,291,456
    float*  kn    = (float*) (ws + 64487424);             //  6,291,456
    float*  vcb   = (float*) (ws + 70778880);             // 12,582,912
    float*  alpha = (float*) (ws + 83361792);             //     98,304
    float*  beta  = (float*) (ws + 83460096);             //     98,304
    float*  osc   = (float*) (ws + 83558400);             // 12,582,912
    bf16_t* og    = (bf16_t*)(ws + 96141312);             //  6,291,456
    bf16_t* woutb = (bf16_t*)(ws + 102432768);            //  4,718,592

    // 1) converts to bf16
    {
        int n;
        n = BT * Dq;          cvt_f32_bf16<<<(n + 255) / 256, 256, 0, stream>>>(x, xb, n);
        n = TQK * Dq;         cvt_f32_bf16<<<(n + 255) / 256, 256, 0, stream>>>(w_q, Wcat, n);
        n = TQK * Dq;         cvt_f32_bf16<<<(n + 255) / 256, 256, 0, stream>>>(w_k, Wcat + (size_t)TQK * Dq, n);
        n = TVq * Dq;         cvt_f32_bf16<<<(n + 255) / 256, 256, 0, stream>>>(w_v, Wcat + (size_t)2 * TQK * Dq, n);
        n = TVq * Dq;         cvt_f32_bf16<<<(n + 255) / 256, 256, 0, stream>>>(w_g, Wcat + ((size_t)2 * TQK + TVq) * Dq, n);
        n = Dq * TVq;         cvt_f32_bf16<<<(n + 255) / 256, 256, 0, stream>>>(w_out, woutb, n);
    }

    // 2) fused projection GEMM: Y(2048,4608) = xb(2048,1536) * Wcat(4608,1536)^T
    {
        dim3 grid(NCAT / GBN, BT / GBM);
        gemm_bf16_wmma<<<grid, 256, 0, stream>>>(xb, Wcat, Y, BT, NCAT, Dq);
    }

    // 3) alpha/beta gates (uses raw f32 x for accuracy; tiny GEMV)
    gates_kernel<<<BT * Hq / 8, 256, 0, stream>>>(x, w_a, w_b, A_log, dt_b, alpha, beta);

    // 4) causal conv + SiLU (+ L2 norm for q,k)
    qk_conv_norm<<<BT, 768, 0, stream>>>(Y, q_cw, k_cw, qn, kn);
    v_conv<<<BT, 512, 0, stream>>>(Y, v_cw, vcb);

    // 5) sequential delta-rule scan (24 independent (b,h) chains)
    gdn_scan_kernel<<<Bq * Hq, 128, 0, stream>>>(qn, kn, vcb, alpha, beta, osc);

    // 6) RMS norm + SiLU gate, pack bf16
    rms_gate_kernel<<<BT * Hq / 8, 256, 0, stream>>>(osc, Y, og);

    // 7) output projection: out(2048,1536) = og(2048,1536) * w_out(1536,1536)^T
    {
        dim3 grid(TVq / GBN, BT / GBM);
        gemm_bf16_wmma<<<grid, 256, 0, stream>>>(og, woutb, out, BT, TVq, Dq);
    }
}
